// TreeCompositionLSTM_38689065402503
// MI455X (gfx1250) — matrix operand order repaired and checked by reading, not compiled, not run on hardware
//
#include <hip/hip_runtime.h>
#include <hip/hip_bf16.h>

// ---------------------------------------------------------------------------
// TreeCompositionLSTM on MI455X (gfx1250, wave32).
// All heavy math routed through v_wmma_f32_16x16x32_f16 (f16 in, f32 acc).
// Main GEMM: block tile 128x128, wave tile 32x64 (8 WMMA / K-step / wave).
// Fused gate-pre: block tile 128x64, wave tile 32x32 (dual accumulators).
// ---------------------------------------------------------------------------

typedef __attribute__((ext_vector_type(16))) _Float16 v16h;
typedef __attribute__((ext_vector_type(8)))  float    v8f;

union Frag16 { uint4 q[2]; v16h v; };
union H16u   { _Float16 h; unsigned short u; };

#define NROWS 16384
#define GATES 5
#define MDIM  512
#define KIN   416   // 300 + 50 + 50 padded to multiple of 32

__device__ __forceinline__ unsigned short f2h(float x) {
    H16u t; t.h = (_Float16)x; return t.u;
}
__device__ __forceinline__ float h2f(unsigned short u) {
    H16u t; t.u = u; return (float)t.h;
}
__device__ __forceinline__ float sigmoidf_(float x) {
    return 1.0f / (1.0f + __expf(-x));
}

// ---------------------------------------------------------------------------
// Tiled WMMA accumulation over K:
//   acc[rt*CT+ct] += A[row0+.., kt..] * B[col0+.., kt..]^T
// A: row-major [*, lda] f16 (as ushort); B: row-major weights [Mout, ldb] f16.
// Block = 256 threads = 8 waves arranged 4 (row slabs of 32) x 2 (col slabs of
// CT*16). LDS row stride = 40 halves (80B): every fragment is two 16B ds reads
// matching the ISA 16-bit A(16x32)/B(32x16) lane layouts.
// ---------------------------------------------------------------------------
template <int CT>
__device__ __forceinline__ void gemm_accum(
    const unsigned short* __restrict__ A, int lda,
    const unsigned short* __restrict__ B, int ldb,
    int row0, int col0, int K,
    unsigned short* As, unsigned short* Bs,
    v8f (&acc)[2 * CT])
{
    const int t    = threadIdx.x;
    const int lane = t & 31;
    const int wave = t >> 5;
    const int wr   = wave >> 1;   // 0..3 : 32-row slab
    const int wc   = wave & 1;    // 0..1 : CT*16-col slab
    const int hh   = lane >> 4;   // half-wave
    const int ll   = lane & 15;

    for (int kt = 0; kt < K; kt += 32) {
        // A tile: 128 rows x 32 halves -> 512 uint4, 2 per thread
        #pragma unroll
        for (int i = 0; i < 2; ++i) {
            int idx = t + i * 256;
            int r   = idx >> 2;
            int c8  = (idx & 3) << 3;
            *(uint4*)&As[r * 40 + c8] =
                *(const uint4*)&A[(size_t)(row0 + r) * lda + kt + c8];
        }
        // B tile: (CT*32) rows x 32 halves -> CT*128 uint4, CT/2 per thread
        #pragma unroll
        for (int i = 0; i < CT / 2; ++i) {
            int idx = t + i * 256;
            int r   = idx >> 2;
            int c8  = (idx & 3) << 3;
            *(uint4*)&Bs[r * 40 + c8] =
                *(const uint4*)&B[(size_t)(col0 + r) * ldb + kt + c8];
        }
        __syncthreads();

        Frag16 a[2], b[CT];
        #pragma unroll
        for (int rt = 0; rt < 2; ++rt) {
            // A frag: lane ll row, halves hold K {hh*8..+7} and {16+hh*8..+7}
            int ar = (wr * 32 + rt * 16 + ll) * 40 + hh * 8;
            a[rt].q[0] = *(const uint4*)&As[ar];
            a[rt].q[1] = *(const uint4*)&As[ar + 16];
        }
        #pragma unroll
        for (int ct = 0; ct < CT; ++ct) {
            // B frag: lane ll col, halves hold K {hh*8..+7} and {16+hh*8..+7}
            int br = (wc * (CT * 16) + ct * 16 + ll) * 40 + hh * 8;
            b[ct].q[0] = *(const uint4*)&Bs[br];
            b[ct].q[1] = *(const uint4*)&Bs[br + 16];
        }
        #pragma unroll
        for (int rt = 0; rt < 2; ++rt)
            #pragma unroll
            for (int ct = 0; ct < CT; ++ct)
                acc[rt * CT + ct] = __builtin_amdgcn_wmma_f32_16x16x32_f16(
                    false, a[rt].v, false, b[ct].v, (short)0,
                    acc[rt * CT + ct], false, false);
        __syncthreads();
    }
}

// C[g] = tanh(A[g] * B[g]^T + bias[g]) stored f16.
// Block tile 128x128 (CT=4). grid = (Mout/128, N/128, G)
__global__ __launch_bounds__(256) void gemm_tanh_kernel(
    const unsigned short* __restrict__ A, int lda, size_t strideA,
    const unsigned short* __restrict__ B, int ldb, size_t strideB,
    const float* __restrict__ bias, int strideBias,
    unsigned short* __restrict__ C, int ldc, size_t strideC,
    int K)
{
    __shared__ __align__(16) unsigned short As[128 * 40];
    __shared__ __align__(16) unsigned short Bs[128 * 40];
    const int g = blockIdx.z;
    A    += (size_t)g * strideA;
    B    += (size_t)g * strideB;
    bias += (size_t)g * strideBias;
    C    += (size_t)g * strideC;
    const int row0 = blockIdx.y * 128;
    const int col0 = blockIdx.x * 128;

    const v8f zero = {0.f,0.f,0.f,0.f,0.f,0.f,0.f,0.f};
    v8f acc[8] = {zero, zero, zero, zero, zero, zero, zero, zero};

    gemm_accum<4>(A, lda, B, ldb, row0, col0, K, As, Bs, acc);

    const int lane = threadIdx.x & 31;
    const int wave = threadIdx.x >> 5;
    const int wr = wave >> 1, wc = wave & 1;
    const int hh = lane >> 4, ll = lane & 15;
    #pragma unroll
    for (int rt = 0; rt < 2; ++rt)
        #pragma unroll
        for (int ct = 0; ct < 4; ++ct) {
            v8f a   = acc[rt * 4 + ct];
            int col = col0 + wc * 64 + ct * 16 + ll;
            float b = bias[col];
            #pragma unroll
            for (int j = 0; j < 8; ++j) {
                int row = row0 + wr * 32 + rt * 16 + hh * 8 + j;
                C[(size_t)row * ldc + col] = f2h(tanhf(a[j] + b));
            }
        }
}

// PRE[g] = tanh(H1[g]*W_last[g]^T + b_last[g]) + h_prev*W_hp[g]^T + b_hp[g]
//                                              + k     *W_k[g]^T  + b_k[g]
// Block tile 128x64 (CT=2): two accumulator sets live simultaneously.
__global__ __launch_bounds__(256) void gate_pre_kernel(
    const unsigned short* __restrict__ H1,
    const unsigned short* __restrict__ HP,
    const unsigned short* __restrict__ KK,
    const unsigned short* __restrict__ WL,
    const unsigned short* __restrict__ WHP,
    const unsigned short* __restrict__ WK,
    const float* __restrict__ bl,
    const float* __restrict__ bhp,
    const float* __restrict__ bk,
    unsigned short* __restrict__ PRE)
{
    __shared__ __align__(16) unsigned short As[128 * 40];
    __shared__ __align__(16) unsigned short Bs[64 * 40];
    const int g    = blockIdx.z;
    const int row0 = blockIdx.y * 128;
    const int col0 = blockIdx.x * 64;
    const size_t actStride = (size_t)NROWS * MDIM;
    const size_t wStride   = (size_t)MDIM * MDIM;

    const v8f zero = {0.f,0.f,0.f,0.f,0.f,0.f,0.f,0.f};
    v8f accm[4] = {zero, zero, zero, zero};   // tanh'd MLP head
    v8f accr[4] = {zero, zero, zero, zero};   // linear h_prev + k part

    gemm_accum<2>(H1 + (size_t)g * actStride, MDIM,
                  WL  + (size_t)g * wStride, MDIM,
                  row0, col0, MDIM, As, Bs, accm);
    gemm_accum<2>(HP, MDIM,
                  WHP + (size_t)g * wStride, MDIM,
                  row0, col0, MDIM, As, Bs, accr);
    gemm_accum<2>(KK, MDIM,
                  WK  + (size_t)g * wStride, MDIM,
                  row0, col0, MDIM, As, Bs, accr);

    const int lane = threadIdx.x & 31;
    const int wave = threadIdx.x >> 5;
    const int wr = wave >> 1, wc = wave & 1;
    const int hh = lane >> 4, ll = lane & 15;
    #pragma unroll
    for (int rt = 0; rt < 2; ++rt)
        #pragma unroll
        for (int ct = 0; ct < 2; ++ct) {
            v8f m = accm[rt * 2 + ct];
            v8f r = accr[rt * 2 + ct];
            int col = col0 + wc * 32 + ct * 16 + ll;
            float bL = bl[g * MDIM + col];
            float bR = bhp[g * MDIM + col] + bk[g * MDIM + col];
            #pragma unroll
            for (int j = 0; j < 8; ++j) {
                int row = row0 + wr * 32 + rt * 16 + hh * 8 + j;
                float pre = tanhf(m[j] + bL) + r[j] + bR;
                PRE[(size_t)g * actStride + (size_t)row * MDIM + col] = f2h(pre);
            }
        }
}

// ---------------------------------------------------------------------------
// Pack / convert helpers
// ---------------------------------------------------------------------------
__global__ void pack_xin_kernel(const float* __restrict__ word,
                                const float* __restrict__ tag,
                                const float* __restrict__ rel,
                                unsigned short* __restrict__ xin)
{
    int idx = blockIdx.x * blockDim.x + threadIdx.x;
    const int total = NROWS * KIN;
    if (idx >= total) return;
    int row = idx / KIN, c = idx - row * KIN;
    float v;
    if      (c < 300) v = word[row * 300 + c];
    else if (c < 350) v = tag[row * 50 + (c - 300)];
    else if (c < 400) v = rel[row * 50 + (c - 350)];
    else              v = 0.0f;
    xin[idx] = f2h(v);
}

__global__ void pack_win_kernel(const float* __restrict__ Ww,
                                const float* __restrict__ Wt,
                                const float* __restrict__ Wr,
                                unsigned short* __restrict__ win)
{
    int idx = blockIdx.x * blockDim.x + threadIdx.x;
    const int total = GATES * MDIM * KIN;
    if (idx >= total) return;
    int g = idx / (MDIM * KIN);
    int r = (idx / KIN) % MDIM;
    int c = idx % KIN;
    float v;
    if      (c < 300) v = Ww[((size_t)g * MDIM + r) * 300 + c];
    else if (c < 350) v = Wt[((size_t)g * MDIM + r) * 50 + (c - 300)];
    else if (c < 400) v = Wr[((size_t)g * MDIM + r) * 50 + (c - 350)];
    else              v = 0.0f;
    win[idx] = f2h(v);
}

__global__ void cvt_f32_f16_kernel(const float* __restrict__ in,
                                   unsigned short* __restrict__ out, int n)
{
    int i = blockIdx.x * blockDim.x + threadIdx.x;
    if (i < n) out[i] = f2h(in[i]);
}

__global__ void bias0_kernel(const float* __restrict__ bw,
                             const float* __restrict__ bt,
                             const float* __restrict__ br,
                             float* __restrict__ b0)
{
    int i = blockIdx.x * blockDim.x + threadIdx.x;
    if (i < GATES * MDIM) b0[i] = bw[i] + bt[i] + br[i];
}

// Final gate combine: c = sig(p0)*tanh(p4) + sig(p1)*q + sig(p2)*c_prev ;
// h = sig(p3)*tanh(c). d_out = [h | c] (f32).
__global__ void combine_kernel(const unsigned short* __restrict__ PRE,
                               const float* __restrict__ q,
                               const float* __restrict__ c_prev,
                               float* __restrict__ out)
{
    const size_t NM = (size_t)NROWS * MDIM;
    size_t i = (size_t)blockIdx.x * blockDim.x + threadIdx.x;
    if (i >= NM) return;
    float p0 = h2f(PRE[0 * NM + i]);
    float p1 = h2f(PRE[1 * NM + i]);
    float p2 = h2f(PRE[2 * NM + i]);
    float p3 = h2f(PRE[3 * NM + i]);
    float p4 = h2f(PRE[4 * NM + i]);
    float ig = sigmoidf_(p0);
    float fd = sigmoidf_(p1);
    float fl = sigmoidf_(p2);
    float og = sigmoidf_(p3);
    float u  = tanhf(p4);
    float c  = ig * u + fd * q[i] + fl * c_prev[i];
    float h  = og * tanhf(c);
    out[i]      = h;
    out[NM + i] = c;
}

// ---------------------------------------------------------------------------
extern "C" void kernel_launch(void* const* d_in, const int* in_sizes, int n_in,
                              void* d_out, int out_size, void* d_ws, size_t ws_size,
                              hipStream_t stream)
{
    (void)in_sizes; (void)n_in; (void)out_size; (void)ws_size;
    const float* word   = (const float*)d_in[0];
    const float* tag    = (const float*)d_in[1];
    const float* rel    = (const float*)d_in[2];
    const float* kin    = (const float*)d_in[3];
    const float* qin    = (const float*)d_in[4];
    const float* h_prev = (const float*)d_in[5];
    const float* c_prev = (const float*)d_in[6];
    const float* W_word = (const float*)d_in[7];
    const float* b_word = (const float*)d_in[8];
    const float* W_tag  = (const float*)d_in[9];
    const float* b_tag  = (const float*)d_in[10];
    const float* W_rel  = (const float*)d_in[11];
    const float* b_rel  = (const float*)d_in[12];
    const float* W_hid  = (const float*)d_in[13];
    const float* b_hid  = (const float*)d_in[14];
    const float* W_last = (const float*)d_in[15];
    const float* b_last = (const float*)d_in[16];
    const float* W_hp   = (const float*)d_in[17];
    const float* b_hp   = (const float*)d_in[18];
    const float* W_k    = (const float*)d_in[19];
    const float* b_k    = (const float*)d_in[20];

    // Workspace carve-up (256B aligned slabs)
    char*  base = (char*)d_ws;
    size_t off  = 0;
    auto take = [&](size_t bytes) -> char* {
        char* p = base + off;
        off += (bytes + 255) & ~(size_t)255;
        return p;
    };
    unsigned short* xin   = (unsigned short*)take((size_t)NROWS * KIN * 2);
    unsigned short* hp16  = (unsigned short*)take((size_t)NROWS * MDIM * 2);
    unsigned short* k16   = (unsigned short*)take((size_t)NROWS * MDIM * 2);
    unsigned short* win   = (unsigned short*)take((size_t)GATES * MDIM * KIN * 2);
    unsigned short* whid  = (unsigned short*)take((size_t)GATES * MDIM * MDIM * 2);
    unsigned short* wlast = (unsigned short*)take((size_t)GATES * MDIM * MDIM * 2);
    unsigned short* whp   = (unsigned short*)take((size_t)GATES * MDIM * MDIM * 2);
    unsigned short* wk    = (unsigned short*)take((size_t)GATES * MDIM * MDIM * 2);
    float*          b0    = (float*)take((size_t)GATES * MDIM * 4);
    unsigned short* H0    = (unsigned short*)take((size_t)GATES * NROWS * MDIM * 2);
    unsigned short* H1    = (unsigned short*)take((size_t)GATES * NROWS * MDIM * 2);
    unsigned short* PRE   = H0;  // stage D only reads H1/hp16/k16 -> reuse H0

    const int nAct = NROWS * MDIM;          // 8,388,608
    const int nW   = GATES * MDIM * MDIM;   // 1,310,720

    // 1) conversions / packing
    pack_xin_kernel<<<(NROWS * KIN + 255) / 256, 256, 0, stream>>>(word, tag, rel, xin);
    cvt_f32_f16_kernel<<<(nAct + 255) / 256, 256, 0, stream>>>(h_prev, hp16, nAct);
    cvt_f32_f16_kernel<<<(nAct + 255) / 256, 256, 0, stream>>>(kin, k16, nAct);
    cvt_f32_f16_kernel<<<(nW + 255) / 256, 256, 0, stream>>>(W_hid, whid, nW);
    cvt_f32_f16_kernel<<<(nW + 255) / 256, 256, 0, stream>>>(W_last, wlast, nW);
    cvt_f32_f16_kernel<<<(nW + 255) / 256, 256, 0, stream>>>(W_hp, whp, nW);
    cvt_f32_f16_kernel<<<(nW + 255) / 256, 256, 0, stream>>>(W_k, wk, nW);
    pack_win_kernel<<<(GATES * MDIM * KIN + 255) / 256, 256, 0, stream>>>(W_word, W_tag, W_rel, win);
    bias0_kernel<<<(GATES * MDIM + 255) / 256, 256, 0, stream>>>(b_word, b_tag, b_rel, b0);

    // 2) GEMM stages (grid.z = gate)
    dim3 gg(MDIM / 128, NROWS / 128, GATES);   // 128x128 block tiles
    // H0[g] = tanh(Xin * Win[g]^T + b0[g])      K = 416 (Xin shared across gates)
    gemm_tanh_kernel<<<gg, 256, 0, stream>>>(
        xin, KIN, (size_t)0,
        win, KIN, (size_t)MDIM * KIN,
        b0, MDIM,
        H0, MDIM, (size_t)NROWS * MDIM,
        KIN);
    // H1[g] = tanh(H0[g] * W_hid[g]^T + b_hid[g])   K = 512
    gemm_tanh_kernel<<<gg, 256, 0, stream>>>(
        H0, MDIM, (size_t)NROWS * MDIM,
        whid, MDIM, (size_t)MDIM * MDIM,
        b_hid, MDIM,
        H1, MDIM, (size_t)NROWS * MDIM,
        MDIM);
    // PRE[g] = tanh(H1*W_last^T + b_last) + hp*W_hp^T + b_hp + k*W_k^T + b_k
    dim3 gp(MDIM / 64, NROWS / 128, GATES);    // 128x64 block tiles
    gate_pre_kernel<<<gp, 256, 0, stream>>>(
        H1, hp16, k16, wlast, whp, wk, b_last, b_hp, b_k, PRE);

    // 3) elementwise gate combine -> d_out = [h | c] f32
    combine_kernel<<<(nAct + 255) / 256, 256, 0, stream>>>(PRE, qin, c_prev, (float*)d_out);
}
